// TensHash_76914274337121
// MI455X (gfx1250) — 8-bit matrix operand order repaired and checked by reading, compiled, not run on hardware
//
#include <hip/hip_runtime.h>
#include <stdint.h>

typedef __attribute__((ext_vector_type(16))) int   v16i;
typedef __attribute__((ext_vector_type(8)))  float v8f;
typedef __attribute__((ext_vector_type(4)))  float v4f;

#define HID      256
#define ROUNDS   64
#define NROWS    16384
#define TILE_M   64
#define NTHREADS 128
#define NHTILES  16
#define BBYTES_PER_ROUND 65536

#define FP8_ONE  0x38u

__global__ void prep_pack_a(const float* __restrict__ M, uint8_t* __restrict__ wsA) {
  int idx  = blockIdx.x * 256 + threadIdx.x;
  int j2   = idx & 31;
  int lane = (idx >> 5) & 31;
  int kb   = (idx >> 10) & 3;
  int nt   = (idx >> 12) & 15;
  int r    = idx >> 16;
  int g = j2 >> 2;
  int j = j2 & 3;
  int h = lane >> 4;
  int rowA = lane & 15;
  int K     = ((g >> 2) << 5) + (h << 4) + ((g & 3) << 2) + j;
  int kglob = (kb << 6) + K;
  int hglob = (nt << 4) + rowA;
  float f = M[((size_t)r << 16) + ((size_t)hglob << 8) + kglob];
  uint8_t e = (f != 0.0f) ? (uint8_t)FP8_ONE : (uint8_t)0;
  if (f < 0.0f) e |= 0x80;
  wsA[idx] = e;
}

__device__ __forceinline__ void async_copy_round(const uint8_t* __restrict__ g,
                                                 uint32_t lds_off, int tid) {
#pragma unroll
  for (int it = 0; it < BBYTES_PER_ROUND / (NTHREADS * 16); ++it) {
    uint32_t dst = lds_off + (uint32_t)(tid * 16 + it * (NTHREADS * 16));
    const uint8_t* src = g + tid * 16 + it * (NTHREADS * 16);
    asm volatile("global_load_async_to_lds_b128 %0, %1, off"
                 :: "v"(dst), "v"(src) : "memory");
  }
}

template <bool HI>
__device__ __forceinline__ uint32_t pack2_fp8(float a, float b, uint32_t old) {
#if __has_builtin(__builtin_amdgcn_cvt_pk_fp8_f32)
  return (uint32_t)__builtin_amdgcn_cvt_pk_fp8_f32(a, b, (int)old, HI);
#else
  uint32_t ea = (a != 0.0f) ? (FP8_ONE | (a < 0.0f ? 0x80u : 0u)) : 0u;
  uint32_t eb = (b != 0.0f) ? (FP8_ONE | (b < 0.0f ? 0x80u : 0u)) : 0u;
  uint32_t v = ea | (eb << 8);
  return HI ? ((old & 0x0000FFFFu) | (v << 16)) : ((old & 0xFFFF0000u) | v);
#endif
}

__launch_bounds__(NTHREADS)
__global__ void tenshash_wmma(const int* __restrict__ nonces,
                              const uint8_t* __restrict__ wsA,
                              const float* __restrict__ noise,
                              float* __restrict__ out) {
  __shared__ uint8_t xs[TILE_M][HID];
  __shared__ uint8_t abuf[2][BBYTES_PER_ROUND];

  const int tid  = threadIdx.x;
  const int lane = tid & 31;
  const int wave = tid >> 5;
  const int hh   = lane >> 4;
  const int n    = lane & 15;
  const int blockRow = blockIdx.x * TILE_M;

  async_copy_round(wsA, (uint32_t)(uintptr_t)&abuf[0][0], tid);

  for (int i = tid; i < TILE_M * HID; i += NTHREADS) {
    int row = i >> 8, k = i & 255;
    int by = k >> 3, bit = k & 7;
    int nb = nonces[(size_t)(blockRow + row) * 32 + by];
    xs[row][k] = ((nb >> bit) & 1) ? (uint8_t)FP8_ONE : (uint8_t)0;
  }

  const float* nrow0 = noise + (size_t)(blockRow + wave * 16 + n) * ROUNDS * HID;

  for (int r = 0; r < ROUNDS; ++r) {
    asm volatile("s_wait_asynccnt 0" ::: "memory");
    __syncthreads();
    if (r + 1 < ROUNDS)
      async_copy_round(wsA + (size_t)(r + 1) * BBYTES_PER_ROUND,
                       (uint32_t)(uintptr_t)&abuf[(r + 1) & 1][0], tid);

    v16i bx[2];
    const uint8_t* xrow = &xs[wave * 16 + n][0];
#pragma unroll
    for (int kp = 0; kp < 2; ++kp) {
#pragma unroll
      for (int q = 0; q < 4; ++q) {
        int4 c = *(const int4*)(xrow + kp * 128 + q * 32 + hh * 16);
        bx[kp][q * 4 + 0] = c.x; bx[kp][q * 4 + 1] = c.y;
        bx[kp][q * 4 + 2] = c.z; bx[kp][q * 4 + 3] = c.w;
      }
    }

    const uint8_t* ab = &abuf[r & 1][0];
    const float* nrow = nrow0 + (size_t)r * HID;
    for (int t = 0; t < NHTILES; ++t) {
      v8f acc = {};
#pragma unroll
      for (int kp = 0; kp < 2; ++kp) {
        v16i af;
        const uint8_t* base = ab + (size_t)((t * 4 + kp * 2) * 32 + lane) * 32;
#pragma unroll
        for (int half64 = 0; half64 < 2; ++half64) {
          const int4* ap = (const int4*)(base + half64 * 1024);
          int4 a0 = ap[0], a1 = ap[1];
          af[half64 * 8 + 0] = a0.x; af[half64 * 8 + 1] = a0.y;
          af[half64 * 8 + 2] = a0.z; af[half64 * 8 + 3] = a0.w;
          af[half64 * 8 + 4] = a1.x; af[half64 * 8 + 5] = a1.y;
          af[half64 * 8 + 6] = a1.z; af[half64 * 8 + 7] = a1.w;
        }
        acc = __builtin_amdgcn_wmma_f32_16x16x128_fp8_fp8(af, bx[kp], (short)0,
                                                          acc, false, false);
      }
      int hbase = t * 16 + hh * 8;
      v4f nz0 = __builtin_nontemporal_load((const v4f*)(nrow + hbase));
      v4f nz1 = __builtin_nontemporal_load((const v4f*)(nrow + hbase + 4));
      float m[8];
#pragma unroll
      for (int v = 0; v < 4; ++v) {
        float y0 = acc[v] + nz0[v];
        float y1 = acc[4 + v] + nz1[v];
        m[v]     = __builtin_fmaf(__builtin_truncf(y0 * 0.5f), -2.0f, y0);
        m[4 + v] = __builtin_fmaf(__builtin_truncf(y1 * 0.5f), -2.0f, y1);
      }
      uint32_t lo = pack2_fp8<false>(m[0], m[1], 0u);
      lo          = pack2_fp8<true >(m[2], m[3], lo);
      uint32_t hi = pack2_fp8<false>(m[4], m[5], 0u);
      hi          = pack2_fp8<true >(m[6], m[7], hi);
      uint2 packed; packed.x = lo; packed.y = hi;
      *(uint2*)&xs[wave * 16 + n][hbase] = packed;
    }
  }

  __syncthreads();
  for (int i = tid; i < TILE_M * HID; i += NTHREADS) {
    int row = i >> 8, k = i & 255;
    uint8_t e = xs[row][k];
    float val = (e & 0x7f) ? ((e & 0x80) ? -1.0f : 1.0f) : 0.0f;
    __builtin_nontemporal_store(val, &out[(size_t)(blockRow + row) * HID + k]);
  }
}

extern "C" void kernel_launch(void* const* d_in, const int* in_sizes, int n_in,
                              void* d_out, int out_size, void* d_ws, size_t ws_size,
                              hipStream_t stream) {
  (void)in_sizes; (void)n_in; (void)out_size; (void)ws_size;
  const int*   nonces   = (const int*)d_in[0];
  const float* matrices = (const float*)d_in[1];
  const float* noise    = (const float*)d_in[2];
  float* out = (float*)d_out;
  uint8_t* wsA = (uint8_t*)d_ws;

  prep_pack_a<<<(ROUNDS * BBYTES_PER_ROUND) / 256, 256, 0, stream>>>(matrices, wsA);
  tenshash_wmma<<<NROWS / TILE_M, NTHREADS, 0, stream>>>(nonces, wsA, noise, out);
}
